// TriangleAttentionStartingNode_40501541601303
// MI455X (gfx1250) — matrix operand order repaired and checked by reading, not compiled, not run on hardware
//
#include <hip/hip_runtime.h>

#define L 256
#define CZ 128
#define NH 4
#define CH 32

typedef __attribute__((ext_vector_type(16))) __bf16 v16bf;
typedef __attribute__((ext_vector_type(8)))  float  v8f;
typedef unsigned short u16;
typedef unsigned int   u32;

union BFV { uint4 q[2]; v16bf v; };

__device__ __forceinline__ u16 f2bf(float f) {
  u32 u = __float_as_uint(f);
  u32 r = u + 0x7FFFu + ((u >> 16) & 1u);
  return (u16)(r >> 16);
}
__device__ __forceinline__ float bf2f(u16 h) {
  return __uint_as_float(((u32)h) << 16);
}

// ---------------------------------------------------------------------------
// K0: weight prep — transpose + f32->bf16 for Wq,Wk,Wv,Wg,Wo. WT[sel][n][k].
// Fold 1/sqrt(C) into Wq.
// ---------------------------------------------------------------------------
__global__ void k_wprep(const float* __restrict__ Wq, const float* __restrict__ Wk,
                        const float* __restrict__ Wv, const float* __restrict__ Wg,
                        const float* __restrict__ Wo, u16* __restrict__ WT) {
  int idx = blockIdx.x * 256 + threadIdx.x;   // 81920 total
  int sel = idx >> 14;
  int e = idx & 16383;
  int k = e >> 7, n = e & 127;
  const float* W = sel == 0 ? Wq : sel == 1 ? Wk : sel == 2 ? Wv : sel == 3 ? Wg : Wo;
  float scale = (sel == 0) ? 0.17677669529663687f : 1.0f;
  WT[sel * 16384 + n * 128 + k] = f2bf(W[k * 128 + n] * scale);
}

// ---------------------------------------------------------------------------
// K1: LayerNorm (one wave per 128-ch row) -> bf16 zn, plus bias = zn @ Wb.
// ---------------------------------------------------------------------------
__global__ void __launch_bounds__(256) k_ln_bias(const float* __restrict__ z,
      const float* __restrict__ gamma, const float* __restrict__ beta,
      const float* __restrict__ Wb, u16* __restrict__ zn, float* __restrict__ biasv) {
  int w = threadIdx.x >> 5, l = threadIdx.x & 31;
  int row = blockIdx.x * 8 + w;
  float4 x = ((const float4*)(z + (size_t)row * CZ))[l];
  float s  = x.x + x.y + x.z + x.w;
  float s2 = x.x * x.x + x.y * x.y + x.z * x.z + x.w * x.w;
#pragma unroll
  for (int o = 1; o < 32; o <<= 1) { s += __shfl_xor(s, o, 32); s2 += __shfl_xor(s2, o, 32); }
  float mu  = s * (1.0f / 128.0f);
  float var = s2 * (1.0f / 128.0f) - mu * mu;
  float rs  = rsqrtf(var + 1e-5f);
  float4 g4 = ((const float4*)gamma)[l];
  float4 b4 = ((const float4*)beta)[l];
  float n0 = (x.x - mu) * rs * g4.x + b4.x;
  float n1 = (x.y - mu) * rs * g4.y + b4.y;
  float n2 = (x.z - mu) * rs * g4.z + b4.z;
  float n3 = (x.w - mu) * rs * g4.w + b4.w;
  u32 p0 = (u32)f2bf(n0) | ((u32)f2bf(n1) << 16);
  u32 p1 = (u32)f2bf(n2) | ((u32)f2bf(n3) << 16);
  ((uint2*)(zn + (size_t)row * CZ))[l] = make_uint2(p0, p1);
  int c0 = l * 4;
#pragma unroll
  for (int hh = 0; hh < 4; hh++) {
    float d = n0 * Wb[(c0 + 0) * 4 + hh] + n1 * Wb[(c0 + 1) * 4 + hh] +
              n2 * Wb[(c0 + 2) * 4 + hh] + n3 * Wb[(c0 + 3) * 4 + hh];
#pragma unroll
    for (int o = 1; o < 32; o <<= 1) d += __shfl_xor(d, o, 32);
    if (l == 0) biasv[row * 4 + hh] = d;
  }
}

// ---------------------------------------------------------------------------
// K2: projections  zn @ {Wq,Wk,Wv,Wg}  via WMMA bf16.
// Block = 1 M-tile (16 rows); A-tile staged once in LDS (4KB, contiguous).
// 8 waves: wave w -> kind=w>>1, ntiles (w&1)*4..+3.
// ---------------------------------------------------------------------------
__global__ void __launch_bounds__(256) k_proj(const u16* __restrict__ zn,
      const u16* __restrict__ WT, const float* __restrict__ bg,
      u16* __restrict__ qo, u16* __restrict__ ko, u16* __restrict__ vo,
      u16* __restrict__ go) {
  __shared__ u16 znT[16 * CZ];
  int w = threadIdx.x >> 5, l = threadIdx.x & 31, lm = l & 15, hf = l >> 4;
  int mtile = blockIdx.x;
  ((uint4*)znT)[threadIdx.x] =
      ((const uint4*)(zn + (size_t)mtile * 16 * CZ))[threadIdx.x];
  __syncthreads();
  int kind = w >> 1;
  int ntb = (w & 1) * 4;
  const u16* WTk = WT + kind * 16384;
  v8f acc[4] = {};
#pragma unroll
  for (int kc = 0; kc < 4; kc++) {
    BFV a;
    const uint4* ap = (const uint4*)(znT + lm * CZ + kc * 32);
    a.q[0] = ap[2 * hf];
    a.q[1] = ap[2 * hf + 1];
#pragma unroll
    for (int t = 0; t < 4; t++) {
      int n = (ntb + t) * 16 + lm;
      BFV b;
      const uint4* bp = (const uint4*)(WTk + n * CZ + kc * 32 + hf * 16);
      b.q[0] = bp[0]; b.q[1] = bp[1];
      acc[t] = __builtin_amdgcn_wmma_f32_16x16x32_bf16(false, a.v, false, b.v,
                                                       (short)0, acc[t], false, false);
    }
  }
  u16* outp = kind == 0 ? qo : kind == 1 ? ko : kind == 2 ? vo : go;
#pragma unroll
  for (int t = 0; t < 4; t++) {
    int n0 = (ntb + t) * 16;
#pragma unroll
    for (int r = 0; r < 8; r++) {
      int rr = mtile * 16 + r + hf * 8;
      int col = n0 + lm;
      float val = acc[t][r];
      if (kind == 3) val = 1.0f / (1.0f + __expf(-(val + bg[col])));
      outp[(size_t)rr * CZ + col] = f2bf(val);
    }
  }
}

// ---------------------------------------------------------------------------
// K3: fused attention per (i,h). 8 waves; wave w does j-strips w and w+8.
// K, V^T, bias, mask staged once in LDS; logits live in 16 WMMA accumulators;
// softmax via half-wave shfl; attn staged bf16 in LDS; attn @ V via WMMA.
// Dynamic LDS: VT 16K + KL 16K + AT 64K + bias 1K + mask 1K = 100352 B.
// ---------------------------------------------------------------------------
__global__ void __launch_bounds__(256) k_attn(const u16* __restrict__ qws,
      const u16* __restrict__ kws, const u16* __restrict__ vws,
      const u16* __restrict__ gws, const float* __restrict__ biasv,
      const int* __restrict__ pm, u16* __restrict__ ows) {
  extern __shared__ char smemraw[];
  u16*   VT    = (u16*)smemraw;          // [32][256]  V transposed
  u16*   KL    = VT + 32 * 256;          // [256][32]  K rows
  u16*   AT    = KL + 256 * 32;          // 8 x [16][256] attn strips
  float* biasL = (float*)(AT + 8 * 16 * 256);
  int*   maskL = (int*)(biasL + 256);

  int tid = threadIdx.x;
  int w = tid >> 5, l = tid & 31, lm = l & 15, hf = l >> 4;
  int i = blockIdx.x >> 2, h = blockIdx.x & 3;
  const int base_ik = i * L;
  const u16* vbase = vws + (size_t)base_ik * CZ + h * CH;
  const u16* kbase = kws + (size_t)base_ik * CZ + h * CH;

  // stage K (contiguous b128 copies) and V (b128 load, transpose scatter)
#pragma unroll
  for (int it = 0; it < 4; it++) {
    int qc = tid + it * 256;            // 1024 chunks of 8 elems
    int kk = qc >> 2, c0 = (qc & 3) * 8;
    uint4 kv = *(const uint4*)(kbase + (size_t)kk * CZ + c0);
    *(uint4*)(KL + kk * 32 + c0) = kv;
    uint4 vv = *(const uint4*)(vbase + (size_t)kk * CZ + c0);
    u32 wd[4] = {vv.x, vv.y, vv.z, vv.w};
#pragma unroll
    for (int j = 0; j < 8; j++) {
      VT[(c0 + j) * 256 + kk] = (u16)((wd[j >> 1] >> ((j & 1) * 16)) & 0xFFFFu);
    }
  }
  biasL[tid] = biasv[(base_ik + tid) * 4 + h];
  maskL[tid] = pm[base_ik + tid];
  __syncthreads();

  u16* myAT = AT + w * 16 * 256;

  for (int s = 0; s < 2; s++) {
    int jbase = (w + s * 8) * 16;
    BFV qa;
    {
      const uint4* ap = (const uint4*)(qws + (size_t)(base_ik + jbase + lm) * CZ + h * CH);
      qa.q[0] = ap[2 * hf];
      qa.q[1] = ap[2 * hf + 1];
    }
    // ---- pass 1: logits = q @ k^T (+bias, mask) ----
    v8f acc[16];
#pragma unroll
    for (int t = 0; t < 16; t++) {
      int kk = t * 16 + lm;
      BFV b;
      const uint4* bp = (const uint4*)(KL + kk * 32 + hf * 16);
      b.q[0] = bp[0]; b.q[1] = bp[1];
      v8f zero = {};
      acc[t] = __builtin_amdgcn_wmma_f32_16x16x32_bf16(false, qa.v, false, b.v,
                                                       (short)0, zero, false, false);
      float bv = biasL[kk];
      bool dead = (maskL[kk] == 0);
#pragma unroll
      for (int r = 0; r < 8; r++) {
        float x = acc[t][r] + bv;
        acc[t][r] = dead ? -1.0e9f : x;
      }
    }
    // ---- softmax over k (lanes 0-15 = row r, lanes 16-31 = row r+8) ----
    float inv[8];
#pragma unroll
    for (int r = 0; r < 8; r++) {
      float m = acc[0][r];
#pragma unroll
      for (int t = 1; t < 16; t++) m = fmaxf(m, acc[t][r]);
#pragma unroll
      for (int o = 1; o < 16; o <<= 1) m = fmaxf(m, __shfl_xor(m, o, 32));
      float ssum = 0.f;
#pragma unroll
      for (int t = 0; t < 16; t++) { float e = __expf(acc[t][r] - m); acc[t][r] = e; ssum += e; }
#pragma unroll
      for (int o = 1; o < 16; o <<= 1) ssum += __shfl_xor(ssum, o, 32);
      inv[r] = 1.0f / ssum;
    }
    // ---- stash attn (bf16, row-major 16x256) in LDS ----
#pragma unroll
    for (int t = 0; t < 16; t++) {
      int col = t * 16 + lm;
#pragma unroll
      for (int r = 0; r < 8; r++) {
        myAT[(r + hf * 8) * 256 + col] = f2bf(acc[t][r] * inv[r]);
      }
    }
    // ---- pass 2: out = attn @ V  (A from LDS, B from VT in LDS) ----
    v8f accO[2] = {};
#pragma unroll
    for (int kc = 0; kc < 8; kc++) {
      BFV a;
      const uint4* ap = (const uint4*)(myAT + lm * 256 + kc * 32);
      a.q[0] = ap[2 * hf];
      a.q[1] = ap[2 * hf + 1];
#pragma unroll
      for (int ch = 0; ch < 2; ch++) {
        int c = ch * 16 + lm;
        BFV b;
        const uint4* bp = (const uint4*)(VT + c * 256 + kc * 32 + hf * 16);
        b.q[0] = bp[0]; b.q[1] = bp[1];
        accO[ch] = __builtin_amdgcn_wmma_f32_16x16x32_bf16(false, a.v, false, b.v,
                                                           (short)0, accO[ch], false, false);
      }
    }
    // ---- gated epilogue ----
#pragma unroll
    for (int ch = 0; ch < 2; ch++) {
#pragma unroll
      for (int r = 0; r < 8; r++) {
        int j = jbase + r + hf * 8;
        int c = ch * 16 + lm;
        size_t idx = (size_t)(base_ik + j) * CZ + h * CH + c;
        float gg = bf2f(gws[idx]);
        ows[idx] = f2bf(accO[ch][r] * gg);
      }
    }
  }
}

// ---------------------------------------------------------------------------
// K4: out = (gated attn_out) @ Wo + bo, masked by pair_mask -> f32 d_out.
// Block = 1 M-tile; A-tile staged in LDS; wave w -> ntile w.
// ---------------------------------------------------------------------------
__global__ void __launch_bounds__(256) k_out(const u16* __restrict__ x,
      const u16* __restrict__ WoT, const float* __restrict__ bo,
      const int* __restrict__ pm, float* __restrict__ out) {
  __shared__ u16 xT[16 * CZ];
  int w = threadIdx.x >> 5, l = threadIdx.x & 31, lm = l & 15, hf = l >> 4;
  int mtile = blockIdx.x;
  ((uint4*)xT)[threadIdx.x] =
      ((const uint4*)(x + (size_t)mtile * 16 * CZ))[threadIdx.x];
  __syncthreads();
  v8f acc = {};
#pragma unroll
  for (int kc = 0; kc < 4; kc++) {
    BFV a;
    const uint4* ap = (const uint4*)(xT + lm * CZ + kc * 32);
    a.q[0] = ap[2 * hf]; a.q[1] = ap[2 * hf + 1];
    BFV b;
    int n = w * 16 + lm;
    const uint4* bp = (const uint4*)(WoT + n * CZ + kc * 32 + hf * 16);
    b.q[0] = bp[0]; b.q[1] = bp[1];
    acc = __builtin_amdgcn_wmma_f32_16x16x32_bf16(false, a.v, false, b.v,
                                                  (short)0, acc, false, false);
  }
#pragma unroll
  for (int r = 0; r < 8; r++) {
    int rr = mtile * 16 + r + hf * 8;
    int col = w * 16 + lm;
    float m = (float)pm[rr];
    out[(size_t)rr * CZ + col] = (acc[r] + bo[col]) * m;
  }
}

// ---------------------------------------------------------------------------
extern "C" void kernel_launch(void* const* d_in, const int* in_sizes, int n_in,
                              void* d_out, int out_size, void* d_ws, size_t ws_size,
                              hipStream_t stream) {
  (void)in_sizes; (void)n_in; (void)out_size; (void)ws_size;
  const float* z     = (const float*)d_in[0];
  const int*   pmask = (const int*)d_in[1];
  const float* gamma = (const float*)d_in[2];
  const float* beta  = (const float*)d_in[3];
  const float* Wq    = (const float*)d_in[4];
  const float* Wk    = (const float*)d_in[5];
  const float* Wv    = (const float*)d_in[6];
  const float* Wb    = (const float*)d_in[7];
  const float* Wg    = (const float*)d_in[8];
  const float* bg    = (const float*)d_in[9];
  const float* Wo    = (const float*)d_in[10];
  const float* bo    = (const float*)d_in[11];
  float* out = (float*)d_out;

  char* ws = (char*)d_ws;
  const size_t NTOK = (size_t)L * L;   // 65536
  const size_t MAT  = NTOK * CZ;       // 8388608 elems
  size_t off = 0;
  auto take = [&](size_t bytes) { size_t o = off; off += (bytes + 255) & ~(size_t)255; return o; };
  u16* zn    = (u16*)(ws + take(MAT * 2));
  u16* qws   = (u16*)(ws + take(MAT * 2));
  u16* kws   = (u16*)(ws + take(MAT * 2));
  u16* vws   = (u16*)(ws + take(MAT * 2));
  u16* gws   = (u16*)(ws + take(MAT * 2));
  u16* ows   = (u16*)(ws + take(MAT * 2));
  float* bia = (float*)(ws + take(NTOK * 4 * sizeof(float)));
  u16* WT    = (u16*)(ws + take(5 * 128 * 128 * 2));

  const int attn_lds = (32 * 256 + 256 * 32 + 8 * 16 * 256) * 2 + 256 * 4 + 256 * 4;

  hipFuncSetAttribute(reinterpret_cast<const void*>(k_attn),
                      hipFuncAttributeMaxDynamicSharedMemorySize, attn_lds);

  k_wprep  <<<320,  256, 0,        stream>>>(Wq, Wk, Wv, Wg, Wo, WT);
  k_ln_bias<<<8192, 256, 0,        stream>>>(z, gamma, beta, Wb, zn, bia);
  k_proj   <<<4096, 256, 0,        stream>>>(zn, WT, bg, qws, kws, vws, gws);
  k_attn   <<<1024, 256, attn_lds, stream>>>(qws, kws, vws, gws, bia, pmask, ows);
  k_out    <<<4096, 256, 0,        stream>>>(ows, WT + 4 * 16384, bo, pmask, out);
}
